// InterlacedSparseAttention_8100308320690
// MI455X (gfx1250) — matrix operand order repaired and checked by reading, not compiled, not run on hardware
//
#include <hip/hip_runtime.h>

typedef __bf16 bf16;
typedef float  v8f   __attribute__((ext_vector_type(8)));
typedef __bf16 v16bf __attribute__((ext_vector_type(16)));

// ---------- helpers ----------
__device__ inline bf16 f2bf(float f) {
    unsigned u = __builtin_bit_cast(unsigned, f);
    unsigned r = (u + 0x7fffu + ((u >> 16) & 1u)) >> 16;
    unsigned short s = (unsigned short)r;
    return __builtin_bit_cast(bf16, s);
}

union BF16x16 { uint4 u[2]; v16bf v; };

// two contiguous 16-byte runs -> one 16-element bf16 fragment
__device__ inline v16bf ldfrag(const bf16* p0, const bf16* p1) {
    BF16x16 t;
    t.u[0] = *reinterpret_cast<const uint4*>(p0);
    t.u[1] = *reinterpret_cast<const uint4*>(p1);
    return t.v;
}

__device__ inline v8f vzero() {
    v8f z = {0.f,0.f,0.f,0.f,0.f,0.f,0.f,0.f};
    return z;
}

__device__ inline v8f wmma_bf(v16bf a, v16bf b, v8f c) {
    return __builtin_amdgcn_wmma_f32_16x16x32_bf16(false, a, false, b, (short)0, c, false, false);
}

// ---------- weight / param prep (f32 -> bf16, epilogue params) ----------
__global__ __launch_bounds__(256)
void prep_k(const float* __restrict__ Wk, const float* __restrict__ bk,
            const float* __restrict__ gk, const float* __restrict__ bnk,
            const float* __restrict__ Wv, const float* __restrict__ bv,
            const float* __restrict__ Ww, const float* __restrict__ bw,
            const float* __restrict__ Wf, const float* __restrict__ bfv,
            const float* __restrict__ gf, const float* __restrict__ bnf,
            bf16* __restrict__ Wkvb, bf16* __restrict__ Wwb, bf16* __restrict__ Wfb,
            float* __restrict__ Pk, float* __restrict__ Pw, float* __restrict__ Pf) {
    int i = blockIdx.x * 256 + threadIdx.x;            // grid covers 512*1024
    if (i < 512 * 512) {
        int o = i >> 9, c = i & 511;
        Wkvb[i] = f2bf(o < 256 ? Wk[o * 512 + c] : Wv[(o - 256) * 512 + c]);
    }
    if (i < 512 * 256) Wwb[i] = f2bf(Ww[i]);
    if (i < 512 * 1024) Wfb[i] = f2bf(Wf[i]);
    if (i < 512) {
        Pk[i]        = i < 256 ? bk[i]  : bv[i - 256];
        Pk[512 + i]  = i < 256 ? gk[i]  : 1.f;
        Pk[1024 + i] = i < 256 ? bnk[i] : 0.f;
        Pk[1536 + i] = i < 256 ? 0.01f  : 1.f;        // slope=1 -> identity
        Pw[i] = bw[i];  Pw[512 + i] = 1.f;   Pw[1024 + i] = 0.f;    Pw[1536 + i] = 1.f;
        Pf[i] = bfv[i]; Pf[512 + i] = gf[i]; Pf[1024 + i] = bnf[i]; Pf[1536 + i] = 0.01f;
    }
}

// ---------- x (NCHW f32) -> pixel-major bf16 ----------
__global__ __launch_bounds__(256)
void x2b_k(const float* __restrict__ x, bf16* __restrict__ Xb) {
    int p = blockIdx.x * 256 + threadIdx.x;            // 0..73727
    int c = blockIdx.y;                                // 0..511
    int n = (p >= 36864) ? 1 : 0;
    int rem = p - n * 36864;
    Xb[(size_t)p * 512 + c] = f2bf(x[((size_t)(n * 512 + c)) * 36864 + rem]);
}

// ---------- generic tiled WMMA GEMM (1x1 conv), compile-time strides ----------
// out[p][o] = epi( W[o][:] . concat(A1[p][:CA], A2[p][:CB]) ),  O = 512
template <int CA, int CB, int OUTMODE>
__global__ __launch_bounds__(256)
void gemm_k(const bf16* __restrict__ A1, const bf16* __restrict__ A2,
            const bf16* __restrict__ Wt, const float* __restrict__ par,
            void* __restrict__ outp, int Npix) {
    constexpr int Cin  = CA + CB;
    constexpr int Ocnt = 512;
    constexpr int numNt = Ocnt / 64;
    const int lane = threadIdx.x & 31;
    const int wv   = threadIdx.x >> 5;
    const int lr = lane & 15, lh = lane >> 4;
    int wid = blockIdx.x * 8 + wv;
    int pt = wid / numNt, nt = wid - pt * numNt;
    if (pt * 32 >= Npix) return;

    v8f acc[2][4];
#pragma unroll
    for (int ms = 0; ms < 2; ++ms)
#pragma unroll
        for (int ns = 0; ns < 4; ++ns) acc[ms][ns] = vzero();

    const size_t prow0 = (size_t)(pt * 32 + lr);
    const size_t prow1 = prow0 + 16;

    // hoisted, compile-time-stride row pointers (immediate offsets in the loop)
    const bf16* a1r0 = A1 + prow0 * CA + lh * 8;
    const bf16* a1r1 = A1 + prow1 * CA + lh * 8;
    const bf16* wrow[4];
#pragma unroll
    for (int ns = 0; ns < 4; ++ns)
        wrow[ns] = Wt + (size_t)(nt * 64 + ns * 16 + lr) * Cin + lh * 8;

#pragma unroll 4
    for (int k0 = 0; k0 < CA; k0 += 32) {
        v16bf a0 = ldfrag(a1r0 + k0, a1r0 + k0 + 16);
        v16bf a1 = ldfrag(a1r1 + k0, a1r1 + k0 + 16);
#pragma unroll
        for (int ns = 0; ns < 4; ++ns) {
            v16bf b = ldfrag(wrow[ns] + k0, wrow[ns] + k0 + 16);
            acc[0][ns] = wmma_bf(a0, b, acc[0][ns]);
            acc[1][ns] = wmma_bf(a1, b, acc[1][ns]);
        }
    }
    if constexpr (CB > 0) {
        const bf16* a2r0 = A2 + prow0 * CB + lh * 8;
        const bf16* a2r1 = A2 + prow1 * CB + lh * 8;
#pragma unroll 4
        for (int k0 = 0; k0 < CB; k0 += 32) {
            v16bf a0 = ldfrag(a2r0 + k0, a2r0 + k0 + 16);
            v16bf a1 = ldfrag(a2r1 + k0, a2r1 + k0 + 16);
#pragma unroll
            for (int ns = 0; ns < 4; ++ns) {
                v16bf b = ldfrag(wrow[ns] + CA + k0, wrow[ns] + CA + k0 + 16);
                acc[0][ns] = wmma_bf(a0, b, acc[0][ns]);
                acc[1][ns] = wmma_bf(a1, b, acc[1][ns]);
            }
        }
    }

#pragma unroll
    for (int ns = 0; ns < 4; ++ns) {
        int o = nt * 64 + ns * 16 + lr;
        float bia = par[o], gn = par[512 + o], bt = par[1024 + o], sl = par[1536 + o];
#pragma unroll
        for (int ms = 0; ms < 2; ++ms) {
#pragma unroll
            for (int r = 0; r < 8; ++r) {
                int row = pt * 32 + ms * 16 + r + lh * 8;
                float z = acc[ms][ns][r] + bia;
                z = z * gn + bt;
                z = (z < 0.f) ? z * sl : z;
                if constexpr (OUTMODE == 0) {
                    ((bf16*)outp)[(size_t)row * Ocnt + o] = f2bf(z);
                } else {
                    int n = (row >= 36864) ? 1 : 0;
                    int rem = row - n * 36864;
                    ((float*)outp)[((size_t)(n * 512 + o)) * 36864 + rem] = z;
                }
            }
        }
    }
}

// ---------- interlaced group pixel mapping ----------
template <int MODE>
__device__ inline int groupPixel(int g, int i) {
    int n, y, x;
    if constexpr (MODE == 0) { // long-range: group=(n,ph,pw), token=(qh,qw) over 24x24
        n = g >> 6; int ph = (g >> 3) & 7, pw = g & 7;
        int qh = i / 24, qw = i - qh * 24;
        y = qh * 8 + ph; x = qw * 8 + pw;
    } else {                   // short-range: group=(n,qh,qw), token=(ph,pw) over 8x8
        n = g / 576; int r = g - n * 576;
        int qh = r / 24, qw = r - qh * 24;
        int ph = i >> 3, pw = i & 7;
        y = qh * 8 + ph; x = qw * 8 + pw;
    }
    return (n * 192 + y) * 192 + x;
}

// ---------- flash attention over interlaced groups ----------
// KQV: [pix][512] bf16 (cols 0..255 = q/k shared, 256..511 = v). CTX: [pix][256] bf16.
template <int MODE, int GS>
__global__ __launch_bounds__(128)
void attn_k(const bf16* __restrict__ KQV, bf16* __restrict__ CTX) {
    __shared__ __align__(16) bf16 Kt[32][256];   // [key][ch]  16 KB
    __shared__ __align__(16) bf16 Vt[256][40];   // [vc][key]  20 KB (padded, 16B-aligned rows)
    __shared__ __align__(16) bf16 Pl[4][16][32]; // per-wave P tile, 4 KB

    const int tid = threadIdx.x, lane = tid & 31, wv = tid >> 5;
    const int lr = lane & 15, lh = lane >> 4;
    constexpr int qbpg = GS / 64;
    const int g  = blockIdx.x / qbpg;
    const int qb = blockIdx.x - g * qbpg;
    const int qi0 = qb * 64 + wv * 16;

    const int qrow = groupPixel<MODE>(g, qi0 + lr);
    const bf16* qbase = KQV + (size_t)qrow * 512;

    float m[8], l[8];
#pragma unroll
    for (int r = 0; r < 8; ++r) { m[r] = -1e30f; l[r] = 0.f; }
    v8f accO[16];
#pragma unroll
    for (int ns = 0; ns < 16; ++ns) accO[ns] = vzero();

    const float scale = 0.0625f;   // 1/sqrt(256)

    for (int kt = 0; kt < GS; kt += 32) {
        __syncthreads();
        {   // stage K (channel-contig) and V (transposed) tiles
            int key = tid >> 2, quarter = tid & 3;
            int kpix = groupPixel<MODE>(g, kt + key);
            const bf16* src = KQV + (size_t)kpix * 512 + quarter * 64;
#pragma unroll
            for (int e = 0; e < 64; e += 8)
                *reinterpret_cast<uint4*>(&Kt[key][quarter * 64 + e]) =
                    *reinterpret_cast<const uint4*>(src + e);
#pragma unroll
            for (int e = 0; e < 64; e += 8) {
                union { uint4 u; bf16 h[8]; } U;
                U.u = *reinterpret_cast<const uint4*>(src + 256 + e);
#pragma unroll
                for (int j = 0; j < 8; ++j) Vt[quarter * 64 + e + j][key] = U.h[j];
            }
        }
        __syncthreads();

        // S = (Q K^T) for 16 queries x 32 keys
        v8f s0 = vzero(), s1 = vzero();
#pragma unroll
        for (int ks = 0; ks < 8; ++ks) {
            v16bf a = ldfrag(qbase + ks * 32 + lh * 16, qbase + ks * 32 + lh * 16 + 8);
            const bf16* b0p = &Kt[lr][ks * 32 + lh * 16];
            const bf16* b1p = &Kt[16 + lr][ks * 32 + lh * 16];
            v16bf b0 = ldfrag(b0p, b0p + 8);
            v16bf b1 = ldfrag(b1p, b1p + 8);
            s0 = wmma_bf(a, b0, s0);
            s1 = wmma_bf(a, b1, s1);
        }

        // online softmax update (rows r + 8*lh; 16-lane row groups)
#pragma unroll
        for (int r = 0; r < 8; ++r) {
            float x0 = s0[r] * scale, x1 = s1[r] * scale;
            float mx = fmaxf(x0, x1);
            mx = fmaxf(mx, __shfl_xor(mx, 1));
            mx = fmaxf(mx, __shfl_xor(mx, 2));
            mx = fmaxf(mx, __shfl_xor(mx, 4));
            mx = fmaxf(mx, __shfl_xor(mx, 8));
            float mn = fmaxf(m[r], mx);
            float alpha = __expf(m[r] - mn);
            float p0 = __expf(x0 - mn), p1 = __expf(x1 - mn);
            float rs = p0 + p1;
            rs += __shfl_xor(rs, 1); rs += __shfl_xor(rs, 2);
            rs += __shfl_xor(rs, 4); rs += __shfl_xor(rs, 8);
            l[r] = l[r] * alpha + rs;
            m[r] = mn;
#pragma unroll
            for (int ns = 0; ns < 16; ++ns) accO[ns][r] *= alpha;
            int row = r + lh * 8;
            Pl[wv][row][lr]      = f2bf(p0);
            Pl[wv][row][16 + lr] = f2bf(p1);
        }

        // O += P @ V   (K = 32 keys)
        const bf16* pap = &Pl[wv][lr][lh * 16];
        v16bf pa = ldfrag(pap, pap + 8);
#pragma unroll
        for (int ns = 0; ns < 16; ++ns) {
            const bf16* vp = &Vt[ns * 16 + lr][lh * 16];
            v16bf vb = ldfrag(vp, vp + 8);
            accO[ns] = wmma_bf(pa, vb, accO[ns]);
        }
    }

    // normalize and write ctx
#pragma unroll
    for (int r = 0; r < 8; ++r) {
        float inv = 1.f / l[r];
        int qi = qi0 + r + lh * 8;
        int qpix = groupPixel<MODE>(g, qi);
        bf16* dst = CTX + (size_t)qpix * 256;
#pragma unroll
        for (int ns = 0; ns < 16; ++ns)
            dst[ns * 16 + lr] = f2bf(accO[ns][r] * inv);
    }
}

// ---------- launch ----------
extern "C" void kernel_launch(void* const* d_in, const int* in_sizes, int n_in,
                              void* d_out, int out_size, void* d_ws, size_t ws_size,
                              hipStream_t stream) {
    const float* x   = (const float*)d_in[0];
    const float* Wk  = (const float*)d_in[1];
    const float* bk  = (const float*)d_in[2];
    const float* gk  = (const float*)d_in[3];
    const float* bnk = (const float*)d_in[4];
    const float* Wv  = (const float*)d_in[5];
    const float* bv  = (const float*)d_in[6];
    const float* Ww  = (const float*)d_in[7];
    const float* bw  = (const float*)d_in[8];
    const float* Wf  = (const float*)d_in[9];
    const float* bfv = (const float*)d_in[10];
    const float* gf  = (const float*)d_in[11];
    const float* bnf = (const float*)d_in[12];
    (void)in_sizes; (void)n_in; (void)out_size; (void)ws_size;

    const size_t NP = 73728;
    char* w = (char*)d_ws;
    auto alloc = [&](size_t bytes) { char* p = w; w += (bytes + 255) & ~(size_t)255; return p; };
    bf16*  Xb   = (bf16*)alloc(NP * 512 * 2);
    bf16*  Yb   = (bf16*)alloc(NP * 512 * 2);
    bf16*  KQVb = (bf16*)alloc(NP * 512 * 2);
    bf16*  CTXb = (bf16*)alloc(NP * 256 * 2);
    bf16*  C2b  = KQVb;                 // KQV is dead after attention; reuse for ctx2
    bf16*  Wkvb = (bf16*)alloc(512 * 512 * 2);
    bf16*  Wwb  = (bf16*)alloc(512 * 256 * 2);
    bf16*  Wfb  = (bf16*)alloc(512 * 1024 * 2);
    float* Pk   = (float*)alloc(4 * 512 * 4);
    float* Pw   = (float*)alloc(4 * 512 * 4);
    float* Pf   = (float*)alloc(4 * 512 * 4);

    prep_k<<<2048, 256, 0, stream>>>(Wk, bk, gk, bnk, Wv, bv, Ww, bw, Wf, bfv, gf, bnf,
                                     Wkvb, Wwb, Wfb, Pk, Pw, Pf);
    x2b_k<<<dim3(288, 512), 256, 0, stream>>>(x, Xb);

    const int gB = (int)((NP / 32) * 8 / 8);   // 2304 blocks, 8 waves each

    // pass 1: long range (128 groups of 576 tokens)
    gemm_k<512, 0, 0><<<gB, 256, 0, stream>>>(Xb, Xb, Wkvb, Pk, KQVb, (int)NP);
    attn_k<0, 576><<<128 * 9, 128, 0, stream>>>(KQVb, CTXb);
    gemm_k<256, 0, 0><<<gB, 256, 0, stream>>>(CTXb, CTXb, Wwb, Pw, C2b, (int)NP);
    gemm_k<512, 512, 0><<<gB, 256, 0, stream>>>(C2b, Xb, Wfb, Pf, Yb, (int)NP);

    // pass 2: short range (1152 groups of 64 tokens)
    gemm_k<512, 0, 0><<<gB, 256, 0, stream>>>(Yb, Yb, Wkvb, Pk, KQVb, (int)NP);
    attn_k<1, 64><<<1152, 128, 0, stream>>>(KQVb, CTXb);
    gemm_k<256, 0, 0><<<gB, 256, 0, stream>>>(CTXb, CTXb, Wwb, Pw, C2b, (int)NP);
    gemm_k<512, 512, 1><<<gB, 256, 0, stream>>>(C2b, Yb, Wfb, Pf, d_out, (int)NP);
}
